// AttentionModule_6004364279816
// MI455X (gfx1250) — matrix operand order repaired and checked by reading, not compiled, not run on hardware
//
#include <hip/hip_runtime.h>
#include <hip/hip_bf16.h>

// ---------------------------------------------------------------------------
// B=1024, F=512, N=8.  All matmuls as f16 WMMA (v_wmma_f32_16x16x32_f16),
// f32 accumulate.  Conv1 exploits the j-degenerate broadcast input (3 column
// classes -> one [8192x1536]x[1536x1536] GEMM); conv2+mean folds to a GEMM.
//
// GEMM core: workgroup 256 threads = 8 waves, tile 128(M) x 128(N), K-step 32.
// Wave (wr,wc) in 4x2 grid computes 32x64 as 2x4 16x16 frags (8 WMMA / iter).
// Staging uses gfx1250 GLOBAL_LOAD_ASYNC_TO_LDS_B128 (ASYNCcnt) via inline asm.
// ---------------------------------------------------------------------------

typedef __attribute__((ext_vector_type(16))) _Float16 v16h;
typedef __attribute__((ext_vector_type(8)))  _Float16 v8h;
typedef __attribute__((ext_vector_type(8)))  float    v8f;

#define BB 1024
#define FF 512
#define NU 8

// ---- gfx1250 async copy helpers (CDNA5 async-to-LDS path, ASYNCcnt) -------
__device__ __forceinline__ void async_copy_b128(void* lds_dst, const void* gsrc) {
  // Low 32 bits of a flat pointer into LDS are the wave-relative LDS offset.
  unsigned ldsoff = (unsigned)(uintptr_t)lds_dst;
  asm volatile("global_load_async_to_lds_b128 %0, %1, off"
               :: "v"(ldsoff), "v"(gsrc) : "memory");
}
__device__ __forceinline__ void wait_async_zero() {
  asm volatile("s_wait_asynccnt 0x0" ::: "memory");
}

// ------------------------- generic WMMA tile core --------------------------
// AMODE 0: A is row-major [M x K] f16 (stride lda), staged via async-to-LDS.
// AMODE 1: A gathered from va[b][c][i] as im2col patch (K = 1536 = (c,di)).
template <int AMODE>
__device__ __forceinline__ void gemm_tile_core(
    const _Float16* __restrict__ Ag, const _Float16* __restrict__ Bg,
    int K, int lda, int ldb, int rowBase, int colBase,
    _Float16* As /*[128*32]*/, _Float16* Bs /*[32*128]*/, v8f acc[2][4]) {
  const int tid  = threadIdx.x;
  const int lane = tid & 31;
  const int wave = tid >> 5;
  const int wr = wave >> 1, wc = wave & 1;
  const int m = lane & 15, hi = lane >> 4;

#pragma unroll
  for (int fm = 0; fm < 2; ++fm)
#pragma unroll
    for (int fn = 0; fn < 4; ++fn)
      acc[fm][fn] = (v8f){0.f, 0.f, 0.f, 0.f, 0.f, 0.f, 0.f, 0.f};

  for (int k0 = 0; k0 < K; k0 += 32) {
    // ---- stage A tile (128 x 32 halves) ----
    {
      const int r    = tid >> 1;
      const int ccol = (tid & 1) * 16;
      if (AMODE == 0) {
        const _Float16* src = Ag + (size_t)(rowBase + r) * lda + k0 + ccol;
        async_copy_b128(As + r * 32 + ccol,     src);
        async_copy_b128(As + r * 32 + ccol + 8, src + 8);
      } else {
        const int grow = rowBase + r;
        const int b = grow >> 3, i = grow & 7;
        const _Float16* vab = Ag + (size_t)b * (FF * NU);
#pragma unroll
        for (int kk = 0; kk < 16; ++kk) {
          const int k  = k0 + ccol + kk;
          const int c  = k / 3;
          const int di = k - 3 * c;
          const int ii = i + di - 1;
          _Float16 v = (_Float16)0.0f;
          if ((unsigned)ii < 8u) v = vab[c * NU + ii];
          As[r * 32 + ccol + kk] = v;
        }
      }
    }
    // ---- stage B tile (32 x 128 halves) ----
    {
      const int rb = tid >> 3;
      const int cb = (tid & 7) * 16;
      const _Float16* src = Bg + (size_t)(k0 + rb) * ldb + colBase + cb;
      _Float16* dst = Bs + rb * 128 + cb;
      async_copy_b128(dst,     src);
      async_copy_b128(dst + 8, src + 8);
    }
    if (k0 + 32 < K)  // gfx1250: lowers to global_prefetch_b8
      __builtin_prefetch(Bg + (size_t)(k0 + 32) * ldb + colBase, 0, 1);
    wait_async_zero();
    __syncthreads();

    // ---- fragments ----
    v16h afr[2], bfr[4];
#pragma unroll
    for (int fm = 0; fm < 2; ++fm) {
      const _Float16* p = As + (wr * 32 + fm * 16 + m) * 32;
      v8h lo = *(const v8h*)(p + hi * 8);
      v8h hh = *(const v8h*)(p + 16 + hi * 8);
#pragma unroll
      for (int e = 0; e < 8; ++e) { afr[fm][e] = lo[e]; afr[fm][e + 8] = hh[e]; }
    }
#pragma unroll
    for (int fn = 0; fn < 4; ++fn) {
      const _Float16* p = Bs + wc * 64 + fn * 16 + m;
#pragma unroll
      for (int e = 0; e < 8; ++e) { bfr[fn][e] = p[(hi * 8 + e) * 128]; bfr[fn][e + 8] = p[(16 + hi * 8 + e) * 128]; }
    }
#pragma unroll
    for (int fm = 0; fm < 2; ++fm)
#pragma unroll
      for (int fn = 0; fn < 4; ++fn)
        acc[fm][fn] = __builtin_amdgcn_wmma_f32_16x16x32_f16(
            false, afr[fm], false, bfr[fn], (short)0, acc[fm][fn], false, false);
    __syncthreads();
  }
}

// Epilogue: element e of acc[fm][fn] lives at
//   row = rowBase + wr*32 + fm*16 + e + 8*hi,  col = colBase + wc*64 + fn*16 + m
#define EPILOGUE_IDX                                 \
  const int tid  = threadIdx.x;                      \
  const int lane = tid & 31;                         \
  const int wave = tid >> 5;                         \
  const int wr = wave >> 1, wc = wave & 1;           \
  const int m = lane & 15, hi = lane >> 4;

// ------------------------------ prep kernels -------------------------------
__global__ void k_pack_x(const float* __restrict__ vx, const float* __restrict__ ax,
                         _Float16* __restrict__ X) {
  int idx = blockIdx.x * blockDim.x + threadIdx.x;  // BB * 2FF
  int b = idx >> 10, i = idx & 1023;
  float v = (i < FF) ? vx[b * FF + i] : ax[b * FF + (i - FF)];
  X[idx] = (_Float16)v;
}

__global__ void k_tr_w1(const float* __restrict__ W1, _Float16* __restrict__ B1) {
  int idx = blockIdx.x * blockDim.x + threadIdx.x;  // NU * 1024 * FF, f contiguous
  int f = idx & 511, i = (idx >> 9) & 1023, n = idx >> 19;
  B1[idx] = (_Float16)W1[((size_t)n * FF + f) * 1024 + i];
}

__global__ void k_tr_w2(const float* __restrict__ W2, _Float16* __restrict__ B2t) {
  int idx = blockIdx.x * blockDim.x + threadIdx.x;  // NU * FF * FF, g contiguous
  int g = idx & 511, f = (idx >> 9) & 511, n = idx >> 18;
  B2t[idx] = (_Float16)W2[((size_t)n * FF + g) * FF + f];
}

// Fold conv1 over the degenerate j axis into 3 column classes (L, M, R).
__global__ void k_fold1(const float* __restrict__ w, _Float16* __restrict__ Bcat) {
  int idx = blockIdx.x * blockDim.x + threadIdx.x;  // 1536 rows (c,di) * 512 o
  int o = idx & 511, k = idx >> 9;
  int c = k / 3, di = k - 3 * c;
  const float* p = w + (((size_t)o * FF + c) * 3 + di) * 3;  // [dj]
  float s0 = p[0], s1 = p[1], s2 = p[2];
  Bcat[(size_t)k * 1536 + o]        = (_Float16)(s1 + s2);       // j = 0
  Bcat[(size_t)k * 1536 + 512 + o]  = (_Float16)(s0 + s1 + s2);  // j = 1..6
  Bcat[(size_t)k * 1536 + 1024 + o] = (_Float16)(s0 + s1);       // j = 7
}

// Fold conv2 (SAME on 2x2) + spatial mean into one [2048 x 512] matrix.
__global__ void k_fold2(const float* __restrict__ w, _Float16* __restrict__ B2f) {
  int idx = blockIdx.x * blockDim.x + threadIdx.x;  // 2048 rows (c,pos) * 512 o
  int o = idx & 511, k = idx >> 9;
  int c = k >> 2, pos = k & 3;
  const float* p = w + ((size_t)o * FF + c) * 9;  // p[kh*3+kw]
  float v;
  if      (pos == 0) v = p[0] + p[1] + p[3] + p[4];
  else if (pos == 1) v = p[1] + p[2] + p[4] + p[5];
  else if (pos == 2) v = p[3] + p[4] + p[6] + p[7];
  else               v = p[4] + p[5] + p[7] + p[8];
  B2f[(size_t)k * FF + o] = (_Float16)(v * 0.25f);
}

// ------------------------------ GEMM kernels -------------------------------
__global__ void k_gemm1(const _Float16* __restrict__ X, const _Float16* __restrict__ B1,
                        const float* __restrict__ b1, _Float16* __restrict__ H) {
  __shared__ __align__(16) _Float16 As[128 * 32];
  __shared__ __align__(16) _Float16 Bs[32 * 128];
  const int colBase = blockIdx.x * 128, rowBase = blockIdx.y * 128, n = blockIdx.z;
  v8f acc[2][4];
  gemm_tile_core<0>(X, B1 + (size_t)n * 1024 * FF, 1024, 1024, FF, rowBase, colBase, As, Bs, acc);
  EPILOGUE_IDX
  const float* bias = b1 + n * FF;
  _Float16* out = H + (size_t)n * BB * FF;
#pragma unroll
  for (int fm = 0; fm < 2; ++fm)
#pragma unroll
    for (int fn = 0; fn < 4; ++fn)
#pragma unroll
      for (int e = 0; e < 8; ++e) {
        int row = rowBase + wr * 32 + fm * 16 + e + 8 * hi;
        int col = colBase + wc * 64 + fn * 16 + m;
        float v = fmaxf(acc[fm][fn][e] + bias[col], 0.f);
        out[(size_t)row * FF + col] = (_Float16)v;
      }
}

__global__ void k_gemm2(const _Float16* __restrict__ H, const _Float16* __restrict__ B2t,
                        const float* __restrict__ b2, const float* __restrict__ vx,
                        _Float16* __restrict__ va) {
  __shared__ __align__(16) _Float16 As[128 * 32];
  __shared__ __align__(16) _Float16 Bs[32 * 128];
  const int colBase = blockIdx.x * 128, rowBase = blockIdx.y * 128, n = blockIdx.z;
  v8f acc[2][4];
  gemm_tile_core<0>(H + (size_t)n * BB * FF, B2t + (size_t)n * FF * FF,
                    FF, FF, FF, rowBase, colBase, As, Bs, acc);
  EPILOGUE_IDX
  const float* bias = b2 + n * FF;
#pragma unroll
  for (int fm = 0; fm < 2; ++fm)
#pragma unroll
    for (int fn = 0; fn < 4; ++fn)
#pragma unroll
      for (int e = 0; e < 8; ++e) {
        int row = rowBase + wr * 32 + fm * 16 + e + 8 * hi;  // b
        int col = colBase + wc * 64 + fn * 16 + m;           // g
        float att = fmaxf(acc[fm][fn][e] + bias[col], 0.f);
        float v = vx[(size_t)row * FF + col] * att;
        va[((size_t)row * FF + col) * NU + n] = (_Float16)v;
      }
}

__global__ void k_gemm3(const _Float16* __restrict__ va, const _Float16* __restrict__ Bcat,
                        _Float16* __restrict__ Y) {
  __shared__ __align__(16) _Float16 As[128 * 32];
  __shared__ __align__(16) _Float16 Bs[32 * 128];
  const int colBase = blockIdx.x * 128, rowBase = blockIdx.y * 128;
  v8f acc[2][4];
  gemm_tile_core<1>(va, Bcat, 1536, 0, 1536, rowBase, colBase, As, Bs, acc);
  EPILOGUE_IDX
#pragma unroll
  for (int fm = 0; fm < 2; ++fm)
#pragma unroll
    for (int fn = 0; fn < 4; ++fn)
#pragma unroll
      for (int e = 0; e < 8; ++e) {
        int row = rowBase + wr * 32 + fm * 16 + e + 8 * hi;
        int col = colBase + wc * 64 + fn * 16 + m;
        Y[(size_t)row * 1536 + col] = (_Float16)acc[fm][fn][e];  // conv1_b added post-pool
      }
}

// Maxpool over i-groups {0..3}/{4..7} and column classes; + conv1 bias.
__global__ void k_pool(const _Float16* __restrict__ Y, const float* __restrict__ c1b,
                       _Float16* __restrict__ A3) {
  int idx = blockIdx.x * blockDim.x + threadIdx.x;  // BB * FF * 4
  int pj = idx & 1, pi = (idx >> 1) & 1, c = (idx >> 2) & 511, b = idx >> 11;
  int col1 = (pj == 0) ? c : 512 + c;         // L or M
  int col2 = (pj == 0) ? 512 + c : 1024 + c;  // M or R
  float mx = -3.4e38f;
#pragma unroll
  for (int i = 0; i < 4; ++i) {
    size_t row = (size_t)b * 8 + pi * 4 + i;
    mx = fmaxf(mx, fmaxf((float)Y[row * 1536 + col1], (float)Y[row * 1536 + col2]));
  }
  A3[(size_t)b * 2048 + c * 4 + pi * 2 + pj] = (_Float16)(mx + c1b[c]);
}

__global__ void k_gemm4(const _Float16* __restrict__ A3, const _Float16* __restrict__ B2f,
                        const float* __restrict__ c2b, float* __restrict__ out) {
  __shared__ __align__(16) _Float16 As[128 * 32];
  __shared__ __align__(16) _Float16 Bs[32 * 128];
  const int colBase = blockIdx.x * 128, rowBase = blockIdx.y * 128;
  v8f acc[2][4];
  gemm_tile_core<0>(A3, B2f, 2048, 2048, FF, rowBase, colBase, As, Bs, acc);
  EPILOGUE_IDX
#pragma unroll
  for (int fm = 0; fm < 2; ++fm)
#pragma unroll
    for (int fn = 0; fn < 4; ++fn)
#pragma unroll
      for (int e = 0; e < 8; ++e) {
        int row = rowBase + wr * 32 + fm * 16 + e + 8 * hi;
        int col = colBase + wc * 64 + fn * 16 + m;
        out[(size_t)row * FF + col] = fmaxf(acc[fm][fn][e] + c2b[col], 0.f);
      }
}

// ------------------------------- launcher ----------------------------------
extern "C" void kernel_launch(void* const* d_in, const int* in_sizes, int n_in,
                              void* d_out, int out_size, void* d_ws, size_t ws_size,
                              hipStream_t stream) {
  const float* vx  = (const float*)d_in[0];
  const float* ax  = (const float*)d_in[1];
  const float* W1  = (const float*)d_in[2];
  const float* b1  = (const float*)d_in[3];
  const float* W2  = (const float*)d_in[4];
  const float* b2  = (const float*)d_in[5];
  const float* c1w = (const float*)d_in[6];
  const float* c1b = (const float*)d_in[7];
  const float* c2w = (const float*)d_in[8];
  const float* c2b = (const float*)d_in[9];
  float* out = (float*)d_out;
  char* ws = (char*)d_ws;

  size_t off = 0;
  auto take = [&](size_t bytes) { size_t o = off; off = (off + bytes + 255) & ~(size_t)255; return o; };
  _Float16* X    = (_Float16*)(ws + take((size_t)BB * 1024 * 2));          // 2 MB
  _Float16* B1   = (_Float16*)(ws + take((size_t)NU * 1024 * FF * 2));     // 8 MB
  _Float16* H    = (_Float16*)(ws + take((size_t)NU * BB * FF * 2));       // 8 MB
  _Float16* B2t  = (_Float16*)(ws + take((size_t)NU * FF * FF * 2));       // 4 MB
  _Float16* va   = (_Float16*)(ws + take((size_t)BB * FF * NU * 2));       // 8 MB
  _Float16* Bcat = (_Float16*)(ws + take((size_t)1536 * 1536 * 2));        // 4.5 MB
  _Float16* Y    = (_Float16*)(ws + take((size_t)8192 * 1536 * 2));        // 24 MB
  _Float16* A3   = (_Float16*)(ws + take((size_t)BB * 2048 * 2));          // 4 MB
  _Float16* B2f  = (_Float16*)(ws + take((size_t)2048 * FF * 2));          // 2 MB

  // prep
  k_pack_x<<<(BB * 1024) / 256, 256, 0, stream>>>(vx, ax, X);
  k_tr_w1 <<<(NU * 1024 * FF) / 256, 256, 0, stream>>>(W1, B1);
  k_tr_w2 <<<(NU * FF * FF) / 256, 256, 0, stream>>>(W2, B2t);
  k_fold1 <<<(1536 * 512) / 256, 256, 0, stream>>>(c1w, Bcat);
  k_fold2 <<<(2048 * 512) / 256, 256, 0, stream>>>(c2w, B2f);

  // WMMA pipeline
  k_gemm1<<<dim3(FF / 128, BB / 128, NU), 256, 0, stream>>>(X, B1, b1, H);
  k_gemm2<<<dim3(FF / 128, BB / 128, NU), 256, 0, stream>>>(H, B2t, b2, vx, va);
  k_gemm3<<<dim3(1536 / 128, 8192 / 128, 1), 256, 0, stream>>>(va, Bcat, Y);
  k_pool <<<(BB * FF * 4) / 256, 256, 0, stream>>>(Y, c1b, A3);
  k_gemm4<<<dim3(FF / 128, BB / 128, 1), 256, 0, stream>>>(A3, B2f, c2b, out);

  (void)in_sizes; (void)n_in; (void)out_size; (void)ws_size;
}
